// MultiHeadAttention_86122684219413
// MI455X (gfx1250) — matrix operand order repaired and checked by reading, not compiled, not run on hardware
//
#include <hip/hip_runtime.h>
#include <hip/hip_bf16.h>
#include <math.h>

// ---- problem constants ----
#define B_  2048
#define N_  200
#define D_  128
#define H_  4
#define DK_ 32
#define E_  4    // NE+1
#define T_  3    // NN+1

typedef _Float16 h16;
typedef __attribute__((ext_vector_type(16))) _Float16 v16h;
typedef __attribute__((ext_vector_type(8)))  float    v8f;
typedef __attribute__((ext_vector_type(4)))  unsigned int u32x4;
typedef __attribute__((ext_vector_type(4)))  int          i32x4;
typedef __attribute__((ext_vector_type(8)))  int          i32x8;

#if defined(__has_builtin)
#if __has_builtin(__builtin_amdgcn_tensor_load_to_lds)
#define HAVE_TDM 1
#endif
#endif

// ---------------------------------------------------------------------------
// WMMA fragment helpers (layouts per CDNA5 ISA §7.12.2, wave32)
// ---------------------------------------------------------------------------
template <typename F>
__device__ inline v16h load_a_frag(int lane, F elem) {
  v16h a;
  int m  = lane & 15;
  int kb = (lane & 16) ? 8 : 0;
#pragma unroll
  for (int i = 0; i < 8; ++i) {
    int k0 = (i < 4) ? (kb + 2 * i) : (kb + 16 + 2 * (i - 4));
    a[2 * i]     = elem(m, k0);
    a[2 * i + 1] = elem(m, k0 + 1);
  }
  return a;
}

template <typename F>
__device__ inline v16h load_b_frag(int lane, F elem) {
  v16h b;
  int n  = lane & 15;
  int kb = (lane & 16) ? 8 : 0;
#pragma unroll
  for (int i = 0; i < 8; ++i) {
    int k0 = (i < 4) ? (kb + 2 * i) : (kb + 16 + 2 * (i - 4));
    b[2 * i]     = elem(k0, n);
    b[2 * i + 1] = elem(k0 + 1, n);
  }
  return b;
}

__device__ inline v8f wmma16(v16h a, v16h b, v8f c) {
  return __builtin_amdgcn_wmma_f32_16x16x32_f16(false, a, false, b,
                                                (short)0, c, false, false);
}

// Pack a 16x16 f32 accumulator tile through per-wave LDS scratch (256 h16)
// and emit wide b128 global stores instead of 8 scalar b16 stores per lane.
__device__ inline void store_tile_f16(const v8f& c, h16* st, int lane,
                                      h16* g, int ldg) {
  int col = lane & 15;
#pragma unroll
  for (int r = 0; r < 8; ++r)
    st[(r + ((lane >> 4) << 3)) * 16 + col] = (h16)c[r];
  asm volatile("s_wait_dscnt 0x0" ::: "memory");
  int row = lane >> 1, half = lane & 1;
  u32x4 d = *(const u32x4*)(st + row * 16 + half * 8);
  *(u32x4*)(g + (size_t)row * ldg + half * 8) = d;
}

// ---------------------------------------------------------------------------
// Tensor Data Mover: DMA a rows x 128 f32 tile (row-major, stride 128) from
// global into LDS. Descriptor per CDNA5 ISA.
// Issued by wave 0 only; completion via TENSORcnt + block barrier.
// ---------------------------------------------------------------------------
__device__ inline void tdm_issue(float* dstLDS, const float* src, int rows,
                                 int wave) {
#ifdef HAVE_TDM
  if (wave == 0) {
    unsigned lds = (unsigned)(size_t)(void*)dstLDS;
    unsigned long long ga = (unsigned long long)(size_t)src;
    u32x4 g0;
    g0[0] = 1u;                                   // count=1, user mode
    g0[1] = lds;                                  // lds_addr
    g0[2] = (unsigned)ga;                         // global_addr[31:0]
    g0[3] = (unsigned)((ga >> 32) & 0x1ffffffull) // global_addr[56:32]
            | (2u << 30);                         // type=2 (image)
    i32x8 g1;
    g1[0] = (int)(2u << 16);                      // data_size=4B
    g1[1] = (int)(128u << 16);                    // tensor_dim0 = 128
    g1[2] = (int)((unsigned)rows << 16);          // tensor_dim1 = rows
    g1[3] = (int)(128u << 16);                    // tile_dim0 = 128
    g1[4] = rows;                                 // tile_dim1 = rows
    g1[5] = 128;                                  // tensor_dim0_stride = 128
    g1[6] = 0;
    g1[7] = 0;
    i32x4 gz = {0, 0, 0, 0};
#if defined(__clang_major__) && (__clang_major__ >= 23)
    i32x8 gz8 = {0, 0, 0, 0, 0, 0, 0, 0};
    __builtin_amdgcn_tensor_load_to_lds(g0, g1, gz, gz, gz8, 0);
#else
    __builtin_amdgcn_tensor_load_to_lds(g0, g1, gz, gz, 0);
#endif
  }
#else
  (void)dstLDS; (void)src; (void)rows; (void)wave;
#endif
}

__device__ inline void tdm_wait(float* dstLDS, const float* src, int rows,
                                int tid) {
#ifdef HAVE_TDM
  (void)dstLDS; (void)src; (void)rows;
  if ((tid >> 5) == 0) __builtin_amdgcn_s_wait_tensorcnt(0);
#else
  for (int i = tid; i < rows * 128; i += 256) dstLDS[i] = src[i];
#endif
}

// ---------------------------------------------------------------------------
// Kernel A: M[e,h] = Wq[e]_h^T . Wk[e]_h  ->  128x128 f16
// ---------------------------------------------------------------------------
__global__ void k_precompute_M(const float* __restrict__ Wq,
                               const float* __restrict__ Wk,
                               h16* __restrict__ M) {
  __shared__ __align__(16) h16 st[4][256];
  int e = blockIdx.x >> 2, h = blockIdx.x & 3;
  int wave = threadIdx.x >> 5, lane = threadIdx.x & 31;
  const float* wq = Wq + ((size_t)e * 128 + h * 32) * 128;
  const float* wk = Wk + ((size_t)e * 128 + h * 32) * 128;
  h16* out = M + (size_t)(e * H_ + h) * 128 * 128;
  for (int t = wave; t < 64; t += 4) {
    int rt = t >> 3, ct = t & 7;
    v16h a = load_a_frag(lane, [&](int m, int kk) -> h16 {
      return (h16)wq[(size_t)kk * 128 + rt * 16 + m];
    });
    v16h bb = load_b_frag(lane, [&](int kk, int n) -> h16 {
      return (h16)wk[(size_t)kk * 128 + ct * 16 + n];
    });
    v8f c = {};
    c = wmma16(a, bb, c);
    store_tile_f16(c, st[wave], lane, out + (size_t)rt * 16 * 128 + ct * 16, 128);
  }
}

// ---------------------------------------------------------------------------
// Kernel B: U[b, eh, :] = q[b,:] . M[eh]  (2048 x 128 x 128 per eh)
// M panel + q panel staged to LDS (coalesced), frags gathered via DS.
// ---------------------------------------------------------------------------
__global__ void k_compute_U(const float* __restrict__ q,
                            const h16* __restrict__ M,
                            h16* __restrict__ U) {
  __shared__ __align__(16) h16 mt[128 * 128];   // 32KB
  __shared__ __align__(16) h16 qt[64 * 128];    // 16KB
  __shared__ __align__(16) h16 st[8][256];      // 4KB
  int bt = blockIdx.x >> 4, eh = blockIdx.x & 15;
  int tid = threadIdx.x, wave = tid >> 5, lane = tid & 31;
  const h16* m = M + (size_t)eh * 128 * 128;
  const float* qrow = q + (size_t)bt * 64 * 128;
  for (int i = tid; i < 8192; i += 256)
    ((unsigned*)mt)[i] = ((const unsigned*)m)[i];
  for (int i = tid; i < 8192; i += 256) qt[i] = (h16)qrow[i];
  __syncthreads();
  for (int t = wave * 4; t < wave * 4 + 4; ++t) {
    int rt = t >> 3, ct = t & 7;
    v8f c = {};
    for (int kt = 0; kt < 4; ++kt) {
      v16h a = load_a_frag(lane, [&](int mm, int kk) -> h16 {
        return qt[(rt * 16 + mm) * 128 + kt * 32 + kk];
      });
      v16h bb = load_b_frag(lane, [&](int kk, int n) -> h16 {
        return mt[(kt * 32 + kk) * 128 + ct * 16 + n];
      });
      c = wmma16(a, bb, c);
    }
    store_tile_f16(c, st[wave], lane,
                   U + (size_t)(bt * 64 + rt * 16) * 2048 + eh * 128 + ct * 16,
                   2048);
  }
}

// ---------------------------------------------------------------------------
// Kernel C: per-batch. TDM k[b] -> LDS; logits GEMM; TDM v[b] overlapped
// with softmax; S = Asel^T @ V GEMM.  Dynamic LDS = 113920 bytes.
// ---------------------------------------------------------------------------
__global__ void k_attn(const float* __restrict__ kin,
                       const float* __restrict__ vin,
                       const h16* __restrict__ U,
                       const float* __restrict__ rel_pri,
                       const int* __restrict__ etype,
                       const int* __restrict__ utype,
                       const int* __restrict__ mask,
                       h16* __restrict__ S,
                       float* __restrict__ attn_out) {
  extern __shared__ __align__(16) char smem[];
  float* kv = (float*)smem;                   // 200*128 f32 = 102400
  h16*   us = (h16*)(smem + 102400);          // 2048 h16   = 4096
  float* lb = (float*)(smem + 106496);        // 4*208 f32  = 3328
  h16*   st = (h16*)(smem + 109824);          // 8*256 h16  = 4096

  int b = blockIdx.x;
  int tid = threadIdx.x, wave = tid >> 5, lane = tid & 31;
  int tu = utype[b];
  const int*   et  = etype + (size_t)b * N_;
  const int*   mk  = mask  + (size_t)b * N_;
  const float* kb_ = kin + (size_t)b * N_ * D_;
  const float* vb_ = vin + (size_t)b * N_ * D_;

  tdm_issue(kv, kb_, N_, wave);                     // DMA k[b] -> LDS
  for (int i = tid; i < 2048; i += 256) us[i] = U[(size_t)b * 2048 + i];
  if (tid < 32) lb[(tid >> 3) * 208 + 200 + (tid & 7)] = -INFINITY;
  tdm_wait(kv, kb_, N_, tid);
  __syncthreads();

  // logits_all[n, eh] = k[b,n,:] . U[b,eh,:]
  for (int rt = wave; rt < 13; rt += 8) {
    v8f c = {};
    for (int kt = 0; kt < 4; ++kt) {
      v16h a = load_a_frag(lane, [&](int m, int kk) -> h16 {
        int n = rt * 16 + m;
        return (n < N_) ? (h16)kv[n * D_ + kt * 32 + kk] : (h16)0.f;
      });
      v16h bb = load_b_frag(lane, [&](int kk, int nn) -> h16 {
        return us[nn * 128 + kt * 32 + kk];
      });
      c = wmma16(a, bb, c);
    }
    int cidx = lane & 15, e = cidx >> 2, h = cidx & 3;
#pragma unroll
    for (int r = 0; r < 8; ++r) {
      int n = rt * 16 + r + ((lane >> 4) << 3);
      if (n < N_ && et[n] == e) {
        float pri = rel_pri[(h * T_ + tu) * E_ + e];
        float val = c[r] * pri * 0.17677669529663687f;  // 1/sqrt(DK)
        if (mk[n]) val = -1e10f;
        lb[h * 208 + n] = val;
      }
    }
  }
  __syncthreads();

  tdm_issue(kv, vb_, N_, wave);   // DMA v[b] -> LDS, overlapped with softmax

  // softmax over n, one wave per head
  if (wave < 4) {
    int h = wave;
    float mx = -INFINITY;
    for (int n = lane; n < 208; n += 32) mx = fmaxf(mx, lb[h * 208 + n]);
    for (int o = 16; o; o >>= 1) mx = fmaxf(mx, __shfl_xor(mx, o, 32));
    float sm = 0.f;
    for (int n = lane; n < 208; n += 32) {
      float ev = __expf(lb[h * 208 + n] - mx);
      lb[h * 208 + n] = ev;
      sm += ev;
    }
    for (int o = 16; o; o >>= 1) sm += __shfl_xor(sm, o, 32);
    float inv = 1.f / sm;
    for (int n = lane; n < 208; n += 32) {
      float a = lb[h * 208 + n] * inv;
      lb[h * 208 + n] = a;
      if (n < N_) attn_out[((size_t)h * B_ + b) * N_ + n] = a;
    }
  }
  tdm_wait(kv, vb_, N_, tid);
  __syncthreads();

  // S[b, eh, :] = sum_n Asel[n, eh] * v[b,n,:]
  {
    int ct = wave;
    v8f c = {};
    for (int kt = 0; kt < 7; ++kt) {
      v16h a = load_a_frag(lane, [&](int m, int kk) -> h16 {
        int n = kt * 32 + kk;
        if (n >= N_) return (h16)0.f;
        int e = m >> 2, h = m & 3;
        return (et[n] == e) ? (h16)lb[h * 208 + n] : (h16)0.f;
      });
      v16h bb = load_b_frag(lane, [&](int kk, int nn) -> h16 {
        int n = kt * 32 + kk;
        return (n < N_) ? (h16)kv[n * D_ + ct * 16 + nn] : (h16)0.f;
      });
      c = wmma16(a, bb, c);
    }
    store_tile_f16(c, st + wave * 256, lane,
                   S + (size_t)b * 2048 + ct * 16, 128);
  }
}

// ---------------------------------------------------------------------------
// Kernel D1: out[b, h*32+j] = sum_{e,d} S[b,e,h,d] * Wv[e, h*32+j, d]
// ---------------------------------------------------------------------------
__global__ void k_outproj(const h16* __restrict__ S,
                          const float* __restrict__ Wv,
                          h16* __restrict__ OUT) {
  __shared__ __align__(16) h16 st[8][256];
  int bt = blockIdx.x >> 2, h = blockIdx.x & 3;
  int wave = threadIdx.x >> 5, lane = threadIdx.x & 31;
  int rt = wave >> 1, ct = wave & 1;
  int b0 = bt * 64 + rt * 16;
  v8f c = {};
  for (int kt = 0; kt < 16; ++kt) {
    v16h a = load_a_frag(lane, [&](int m, int kk) -> h16 {
      int K = kt * 32 + kk; int e = K >> 7, d = K & 127;
      return S[(size_t)(b0 + m) * 2048 + (e * 4 + h) * 128 + d];
    });
    v16h bb = load_b_frag(lane, [&](int kk, int nn) -> h16 {
      int K = kt * 32 + kk; int e = K >> 7, d = K & 127;
      int j = ct * 16 + nn;
      return (h16)Wv[((size_t)e * 128 + h * 32 + j) * 128 + d];
    });
    c = wmma16(a, bb, c);
  }
  store_tile_f16(c, st[wave], lane,
                 OUT + (size_t)b0 * 128 + h * 32 + ct * 16, 128);
}

// ---------------------------------------------------------------------------
// Kernel D2: o = out @ W_fc[t(b)] + b_fc; x = o + q; LayerNorm -> y
// ---------------------------------------------------------------------------
__global__ void k_fc_ln(const h16* __restrict__ OUT,
                        const float* __restrict__ W_fc,
                        const float* __restrict__ b_fc,
                        const float* __restrict__ q,
                        const int* __restrict__ utype,
                        const float* __restrict__ gamma,
                        const float* __restrict__ beta,
                        float* __restrict__ y) {
  __shared__ float x[64][128];
  int bt = blockIdx.x;
  int tid = threadIdx.x, wave = tid >> 5, lane = tid & 31;
  for (int t = wave; t < 96; t += 8) {
    int tv = t / 32, rem = t % 32;
    int rt = rem >> 3, ct = rem & 7;
    int b0 = bt * 64 + rt * 16;
    v8f c = {};
    for (int kt = 0; kt < 4; ++kt) {
      v16h a = load_a_frag(lane, [&](int m, int kk) -> h16 {
        return OUT[(size_t)(b0 + m) * 128 + kt * 32 + kk];
      });
      v16h bb = load_b_frag(lane, [&](int kk, int nn) -> h16 {
        return (h16)W_fc[((size_t)tv * 128 + kt * 32 + kk) * 128 + ct * 16 + nn];
      });
      c = wmma16(a, bb, c);
    }
    int col = ct * 16 + (lane & 15);
#pragma unroll
    for (int r = 0; r < 8; ++r) {
      int b = b0 + r + ((lane >> 4) << 3);
      if (utype[b] == tv) {
        x[b - bt * 64][col] =
            c[r] + b_fc[tv * 128 + col] + q[(size_t)b * 128 + col];
      }
    }
  }
  __syncthreads();
  if (tid < 64) {
    int b = bt * 64 + tid;
    float mu = 0.f;
    for (int i = 0; i < 128; ++i) mu += x[tid][i];
    mu *= (1.f / 128.f);
    float var = 0.f;
    for (int i = 0; i < 128; ++i) { float d = x[tid][i] - mu; var += d * d; }
    var *= (1.f / 128.f);
    float inv = rsqrtf(var + 1e-5f);
    for (int i = 0; i < 128; ++i)
      y[(size_t)b * 128 + i] = (x[tid][i] - mu) * inv * gamma[i] + beta[i];
  }
}

// ---------------------------------------------------------------------------
extern "C" void kernel_launch(void* const* d_in, const int* in_sizes, int n_in,
                              void* d_out, int out_size, void* d_ws,
                              size_t ws_size, hipStream_t stream) {
  (void)in_sizes; (void)n_in; (void)out_size; (void)ws_size;
  const float* q       = (const float*)d_in[0];
  const float* k       = (const float*)d_in[1];
  const float* v       = (const float*)d_in[2];
  const float* Wq      = (const float*)d_in[3];
  const float* Wk      = (const float*)d_in[4];
  const float* Wv      = (const float*)d_in[5];
  const float* rel_pri = (const float*)d_in[6];
  const float* W_fc    = (const float*)d_in[7];
  const float* b_fc    = (const float*)d_in[8];
  const float* gamma   = (const float*)d_in[9];
  const float* beta    = (const float*)d_in[10];
  const int*   etype   = (const int*)d_in[11];
  const int*   utype   = (const int*)d_in[12];
  const int*   mask    = (const int*)d_in[13];

  float* y    = (float*)d_out;
  float* attn = y + (size_t)B_ * D_;

  // workspace layout (f16): M 512KB | U 8MB | S 8MB | OUT 512KB
  char* ws = (char*)d_ws;
  h16* M   = (h16*)(ws);
  h16* U   = (h16*)(ws + (512u << 10));
  h16* S   = (h16*)(ws + (512u << 10) + (8u << 20));
  h16* OUT = (h16*)(ws + (512u << 10) + (16u << 20));

  k_precompute_M<<<dim3(16), dim3(128), 0, stream>>>(Wq, Wk, M);
  k_compute_U<<<dim3(512), dim3(256), 0, stream>>>(q, M, U);
  k_attn<<<dim3(B_), dim3(256), 113920, stream>>>(k, v, U, rel_pri, etype,
                                                  utype, mask, S, attn);
  k_outproj<<<dim3(128), dim3(256), 0, stream>>>(S, Wv, OUT);
  k_fc_ln<<<dim3(32), dim3(256), 0, stream>>>(OUT, W_fc, b_fc, q, utype,
                                              gamma, beta, y);
}
